// TransformerLayerXL_3204045603390
// MI455X (gfx1250) — compile-verified, matrix operand order checked by reading, not run on hardware
//
#include <hip/hip_runtime.h>
#include <hip/hip_bf16.h>
#include <stdint.h>

// ---------------------------------------------------------------------------
// Types for CDNA5 WMMA (wave32): A/B = v16h (16 halves / lane), C/D = v8f.
// ---------------------------------------------------------------------------
typedef _Float16 v8h  __attribute__((ext_vector_type(8)));
typedef _Float16 v16h __attribute__((ext_vector_type(16)));
typedef float    v8f  __attribute__((ext_vector_type(8)));

union H16 { v16h v; v8h h[2]; };

static __device__ inline v8f wmma16(const H16& a, const H16& b, v8f c) {
  // D = A(16x32 f16) * B(32x16 f16) + C(16x16 f32)
  return __builtin_amdgcn_wmma_f32_16x16x32_f16(
      /*neg_a=*/false, a.v, /*neg_b=*/false, b.v,
      /*c_mod=*/(short)0, c, /*reuse_a=*/false, /*reuse_b=*/false);
}

// Problem constants
#define Bq   8
#define Sq   512
#define Dq   2048
#define Hq   32
#define HDq  64
#define Iq   8192
#define Mrows (Bq * Sq)        // 4096
#define RELq 1023

// ---------------------------------------------------------------------------
// fp32 -> f16 elementwise convert
// ---------------------------------------------------------------------------
__global__ void cvt_f16_kernel(const float* __restrict__ in, _Float16* __restrict__ out, int n) {
  int i = blockIdx.x * blockDim.x + threadIdx.x;
  if (i < n) out[i] = (_Float16)in[i];
}

// ---------------------------------------------------------------------------
// Weight transpose: W[K][N] fp32 -> Wt[N][K] f16 (32x32 LDS tiles)
// ---------------------------------------------------------------------------
__global__ __launch_bounds__(256) void transpose_w_kernel(const float* __restrict__ W,
                                                          _Float16* __restrict__ Wt,
                                                          int K, int N) {
  __shared__ float t[32][33];
  int nb = blockIdx.x * 32, kb = blockIdx.y * 32;
  int tx = threadIdx.x, ty = threadIdx.y;   // block (32,8)
#pragma unroll
  for (int i = 0; i < 32; i += 8)
    t[ty + i][tx] = W[(size_t)(kb + ty + i) * N + nb + tx];
  __syncthreads();
#pragma unroll
  for (int i = 0; i < 32; i += 8)
    Wt[(size_t)(nb + ty + i) * K + kb + tx] = (_Float16)t[tx][ty + i];
}

// ---------------------------------------------------------------------------
// Relative-position bias row: brow[r] = mean(rel_pos[r][:64])
// ---------------------------------------------------------------------------
__global__ void bias_row_kernel(const float* __restrict__ rel, float* __restrict__ brow) {
  int r = blockIdx.x;
  int lane = threadIdx.x;                           // 32 threads
  float s = rel[r * 64 + lane] + rel[r * 64 + 32 + lane];
#pragma unroll
  for (int m = 16; m >= 1; m >>= 1) s += __shfl_xor(s, m);
  if (lane == 0) brow[r] = s * (1.0f / 64.0f);
}

// ---------------------------------------------------------------------------
// WMMA GEMM: C[M][N] = A[M][K](f16, row-major) * W[K][N], with W given as
// Wt[N][K] (f16). Block = 256 thr = 8 waves; block tile 128(M) x 128(N);
// wave tile 16 x 128; K-step 32.
// B tile is double-buffered in LDS and filled with CDNA5 async-to-LDS copies
// (global_load_async_to_lds_b128, tracked by ASYNCcnt): one barrier/K-step,
// no VGPR round-trip for the weight tile.
// ---------------------------------------------------------------------------
#define BT_STRIDE 40            // 32 K-halves + 8 pad (16B-aligned fragments)
#define BT_BUF    (128 * BT_STRIDE)

template <bool OUT_HALF>
__global__ __launch_bounds__(256) void gemm_wmma_kernel(const _Float16* __restrict__ A,
                                                        const _Float16* __restrict__ Wt,
                                                        float* __restrict__ Cf,
                                                        _Float16* __restrict__ Ch,
                                                        int M, int N, int K) {
  __shared__ _Float16 bt[2 * BT_BUF];
  const int tid  = threadIdx.x;
  const int lane = tid & 31;
  const int wid  = tid >> 5;
  const int lgrp = lane >> 4;   // 0/1: K-subgroup per CDNA5 16-bit layout
  const int lmod = lane & 15;
  const int nb = blockIdx.x * 128;
  const int mb = blockIdx.y * 128;

  const _Float16* arow = A + (size_t)(mb + wid * 16 + lmod) * K;

  // Async fill of one 128(n) x 32(k) B tile: 512 x 16B chunks, 2 per thread.
  auto issue_fill = [&](int buf, int k0) {
#pragma unroll
    for (int c = 0; c < 2; c++) {
      int ch = tid + c * 256;          // 0..511
      int n  = ch >> 2;
      int kq = ch & 3;
      const _Float16* gsrc = Wt + (size_t)(nb + n) * K + k0 + kq * 8;
      unsigned ldst = (unsigned)(uintptr_t)(bt + buf * BT_BUF + n * BT_STRIDE + kq * 8);
      asm volatile("global_load_async_to_lds_b128 %0, %1, off"
                   :: "v"(ldst), "v"(gsrc) : "memory");
    }
  };

  v8f z = {0.f, 0.f, 0.f, 0.f, 0.f, 0.f, 0.f, 0.f};
  v8f acc[8];
#pragma unroll
  for (int j = 0; j < 8; j++) acc[j] = z;

  issue_fill(0, 0);
  asm volatile("s_wait_asynccnt 0x0" ::: "memory");
  __syncthreads();

  const int nsteps = K >> 5;
  for (int s = 0; s < nsteps; s++) {
    const int k0  = s << 5;
    const int cur = s & 1;
    if (s + 1 < nsteps) issue_fill(1 - cur, k0 + 32);
    if (k0 + 64 < K)
      __builtin_prefetch(arow + k0 + 64, 0, 1);

    // A fragment (16x32): lane = row lmod, halves = K {lgrp*8..+7, 16+lgrp*8..+7}
    H16 a;
    a.h[0] = *(const v8h*)(arow + k0 + lgrp * 8);
    a.h[1] = *(const v8h*)(arow + k0 + 16 + lgrp * 8);

    const _Float16* btc = bt + cur * BT_BUF;
#pragma unroll
    for (int j = 0; j < 8; j++) {
      H16 b;
      const _Float16* bp = btc + (j * 16 + lmod) * BT_STRIDE + lgrp * 8;
      b.h[0] = *(const v8h*)bp;
      b.h[1] = *(const v8h*)(bp + 16);
      acc[j] = wmma16(a, b, acc[j]);
    }

    asm volatile("s_wait_asynccnt 0x0" ::: "memory");
    __syncthreads();
  }

  // C layout: vgpr e -> row e (lanes 0-15) / e+8 (lanes 16-31), col = lmod
#pragma unroll
  for (int j = 0; j < 8; j++) {
#pragma unroll
    for (int e = 0; e < 8; e++) {
      int row = mb + wid * 16 + e + lgrp * 8;
      int col = nb + j * 16 + lmod;
      if (OUT_HALF) Ch[(size_t)row * N + col] = (_Float16)acc[j][e];
      else          Cf[(size_t)row * N + col] = acc[j][e];
    }
  }
}

// ---------------------------------------------------------------------------
// Flash-style attention with WMMA. 1 block per (b,h); 4 waves; each wave owns
// a 16-row q strip inside a 64-row q tile; kv streamed in tiles of 32.
// Q/K/V f16 stored [B,S,H,HD] (row stride D). ctx written f16 same layout.
// ---------------------------------------------------------------------------
__global__ __launch_bounds__(128) void attn_kernel(const _Float16* __restrict__ Qh,
                                                   const _Float16* __restrict__ Kh,
                                                   const _Float16* __restrict__ Vh,
                                                   const float* __restrict__ brow,
                                                   _Float16* __restrict__ ctxh) {
  __shared__ _Float16 vt[64 * 40];        // V tile transposed: [hd][kv], stride 40
  __shared__ _Float16 pbuf[4 * 16 * 40];  // per-wave P staging: [16 q][32 kv], stride 40
  __shared__ float bsh[RELq];

  const int bh   = blockIdx.x;
  const int b    = bh >> 5;
  const int h    = bh & (Hq - 1);
  const int tid  = threadIdx.x;
  const int lane = tid & 31;
  const int wid  = tid >> 5;
  const int lgrp = lane >> 4;
  const int lmod = lane & 15;

  for (int i = tid; i < RELq; i += 128) bsh[i] = brow[i];

  const size_t headoff = (size_t)b * Sq * Dq + (size_t)h * HDq;
  const _Float16* Qb = Qh + headoff;
  const _Float16* Kb = Kh + headoff;
  const _Float16* Vb = Vh + headoff;
  _Float16*       Cb = ctxh + headoff;

  v8f z = {0.f, 0.f, 0.f, 0.f, 0.f, 0.f, 0.f, 0.f};

  for (int q0 = 0; q0 < Sq; q0 += 64) {
    const int qw = q0 + wid * 16;  // this wave's q base
    // Q fragments for hd 0..31 and 32..63
    const _Float16* qrow = Qb + (size_t)(qw + lmod) * Dq;
    H16 qa0, qa1;
    qa0.h[0] = *(const v8h*)(qrow + lgrp * 8);
    qa0.h[1] = *(const v8h*)(qrow + 16 + lgrp * 8);
    qa1.h[0] = *(const v8h*)(qrow + 32 + lgrp * 8);
    qa1.h[1] = *(const v8h*)(qrow + 48 + lgrp * 8);

    v8f ctx[4];
#pragma unroll
    for (int j = 0; j < 4; j++) ctx[j] = z;
    float mrow[8], lrow[8];
#pragma unroll
    for (int e = 0; e < 8; e++) { mrow[e] = -1e30f; lrow[e] = 0.f; }

    for (int kv0 = 0; kv0 < Sq; kv0 += 32) {
      __syncthreads();  // protect vt from previous iteration's readers
      // Transpose 32(kv) x 64(hd) V tile into LDS
      for (int i = tid; i < 2048; i += 128) {
        int kvl = i >> 6, hd = i & 63;
        vt[hd * 40 + kvl] = Vb[(size_t)(kv0 + kvl) * Dq + hd];
      }
      __syncthreads();

      // Scores: two 16x16 tiles, each = 2 chained WMMAs over HD=64
      v8f sv[2];
#pragma unroll
      for (int t = 0; t < 2; t++) {
        const _Float16* krow = Kb + (size_t)(kv0 + t * 16 + lmod) * Dq;
        H16 kb0, kb1;
        kb0.h[0] = *(const v8h*)(krow + lgrp * 8);
        kb0.h[1] = *(const v8h*)(krow + 16 + lgrp * 8);
        kb1.h[0] = *(const v8h*)(krow + 32 + lgrp * 8);
        kb1.h[1] = *(const v8h*)(krow + 48 + lgrp * 8);
        v8f a = z;
        a = wmma16(qa0, kb0, a);
        a = wmma16(qa1, kb1, a);
        sv[t] = a;
      }

      // scale + relative-position bias.
      // idx = kv - q + 511 is always within [0,1022] for S=512: no clamp.
#pragma unroll
      for (int t = 0; t < 2; t++) {
        const int ibase = kv0 + t * 16 + lmod - (qw + lgrp * 8) + 511;
#pragma unroll
        for (int e = 0; e < 8; e++) {
          sv[t][e] = sv[t][e] * 0.125f + bsh[ibase - e];
        }
      }

      // Online softmax (row reductions across 16-lane groups)
      float scl[8];
#pragma unroll
      for (int e = 0; e < 8; e++) {
        float tmax = fmaxf(sv[0][e], sv[1][e]);
        tmax = fmaxf(tmax, __shfl_xor(tmax, 1));
        tmax = fmaxf(tmax, __shfl_xor(tmax, 2));
        tmax = fmaxf(tmax, __shfl_xor(tmax, 4));
        tmax = fmaxf(tmax, __shfl_xor(tmax, 8));
        float mn = fmaxf(mrow[e], tmax);
        scl[e] = __expf(mrow[e] - mn);
        mrow[e] = mn;
        float p0 = __expf(sv[0][e] - mn);
        float p1 = __expf(sv[1][e] - mn);
        sv[0][e] = p0; sv[1][e] = p1;
        float rs = p0 + p1;
        rs += __shfl_xor(rs, 1);
        rs += __shfl_xor(rs, 2);
        rs += __shfl_xor(rs, 4);
        rs += __shfl_xor(rs, 8);
        lrow[e] = lrow[e] * scl[e] + rs;
      }
      // rescale running ctx
#pragma unroll
      for (int j = 0; j < 4; j++)
#pragma unroll
        for (int e = 0; e < 8; e++) ctx[j][e] *= scl[e];

      // Reshape P (C layout) -> A fragment through per-wave LDS staging
      _Float16* pb = pbuf + wid * 16 * 40;
#pragma unroll
      for (int t = 0; t < 2; t++)
#pragma unroll
        for (int e = 0; e < 8; e++)
          pb[(e + lgrp * 8) * 40 + t * 16 + lmod] = (_Float16)sv[t][e];

      H16 pa;
      pa.h[0] = *(const v8h*)(pb + lmod * 40 + lgrp * 8);
      pa.h[1] = *(const v8h*)(pb + lmod * 40 + 16 + lgrp * 8);

      // ctx += P(16x32) * V(32x64), hd tiles of 16
#pragma unroll
      for (int j = 0; j < 4; j++) {
        H16 vb;
        const _Float16* vp = vt + (j * 16 + lmod) * 40 + lgrp * 8;
        vb.h[0] = *(const v8h*)vp;
        vb.h[1] = *(const v8h*)(vp + 16);
        ctx[j] = wmma16(pa, vb, ctx[j]);
      }
    }

    // Normalize and store ctx (f16, [B,S,H,HD])
#pragma unroll
    for (int e = 0; e < 8; e++) {
      float inv = 1.0f / lrow[e];
      int qi = qw + e + lgrp * 8;
#pragma unroll
      for (int j = 0; j < 4; j++) {
        int hd = j * 16 + lmod;
        Cb[(size_t)qi * Dq + hd] = (_Float16)(ctx[j][e] * inv);
      }
    }
  }
}

// ---------------------------------------------------------------------------
// out = LayerNorm(x + y) * g + b (fp32), optional f16 side copy. 1 block/row.
// ---------------------------------------------------------------------------
__global__ __launch_bounds__(256) void add_ln_kernel(const float* __restrict__ x,
                                                     const float* __restrict__ y,
                                                     const float* __restrict__ g,
                                                     const float* __restrict__ bb,
                                                     float* __restrict__ out,
                                                     _Float16* __restrict__ outh) {
  __shared__ float red[256];
  const int row = blockIdx.x;
  const int tid = threadIdx.x;
  const float* xr = x + (size_t)row * Dq;
  const float* yr = y + (size_t)row * Dq;

  float v[8];
  float s = 0.f;
#pragma unroll
  for (int i = 0; i < 8; i++) {
    v[i] = xr[tid + i * 256] + yr[tid + i * 256];
    s += v[i];
  }
  red[tid] = s;
  __syncthreads();
  for (int o = 128; o > 0; o >>= 1) {
    if (tid < o) red[tid] += red[tid + o];
    __syncthreads();
  }
  float mean = red[0] * (1.0f / Dq);
  __syncthreads();

  float s2 = 0.f;
#pragma unroll
  for (int i = 0; i < 8; i++) {
    float d = v[i] - mean;
    s2 += d * d;
  }
  red[tid] = s2;
  __syncthreads();
  for (int o = 128; o > 0; o >>= 1) {
    if (tid < o) red[tid] += red[tid + o];
    __syncthreads();
  }
  float inv = rsqrtf(red[0] * (1.0f / Dq) + 1e-6f);

#pragma unroll
  for (int i = 0; i < 8; i++) {
    int c = tid + i * 256;
    float o = (v[i] - mean) * inv * g[c] + bb[c];
    out[(size_t)row * Dq + c] = o;
    if (outh) outh[(size_t)row * Dq + c] = (_Float16)o;
  }
}

// ---------------------------------------------------------------------------
// SwiGLU elementwise: gu = silu(gate) * up   (f16 in/out, fp32 math)
// ---------------------------------------------------------------------------
__global__ void silu_mul_kernel(const _Float16* __restrict__ gate,
                                const _Float16* __restrict__ up,
                                _Float16* __restrict__ gu, int n) {
  int i = blockIdx.x * blockDim.x + threadIdx.x;
  if (i < n) {
    float gx = (float)gate[i];
    float sl = gx / (1.0f + __expf(-gx));
    gu[i] = (_Float16)(sl * (float)up[i]);
  }
}

// ---------------------------------------------------------------------------
// Host-side orchestration
// ---------------------------------------------------------------------------
static constexpr size_t SZ_ACT_H  = (size_t)Mrows * Dq * 2;   // 16.8 MB
static constexpr size_t SZ_ACT_F  = (size_t)Mrows * Dq * 4;   // 33.6 MB
static constexpr size_t SZ_WT     = (size_t)Dq * Iq * 2;      // 33.6 MB (max weight)
static constexpr size_t SZ_BIG_H  = (size_t)Mrows * Iq * 2;   // 67.1 MB

static constexpr size_t OFF_XH   = 0;
static constexpr size_t OFF_WT   = OFF_XH   + SZ_ACT_H;
static constexpr size_t OFF_QH   = OFF_WT   + SZ_WT;
static constexpr size_t OFF_KH   = OFF_QH   + SZ_ACT_H;
static constexpr size_t OFF_VH   = OFF_KH   + SZ_ACT_H;
static constexpr size_t OFF_CTX  = OFF_VH   + SZ_ACT_H;
static constexpr size_t OFF_F32A = OFF_CTX  + SZ_ACT_H;       // attn proj / ffn result
static constexpr size_t OFF_H1   = OFF_F32A + SZ_ACT_F;
static constexpr size_t OFF_H1H  = OFF_H1   + SZ_ACT_F;
static constexpr size_t OFF_GATE = OFF_H1H  + SZ_ACT_H;
static constexpr size_t OFF_UP   = OFF_GATE + SZ_BIG_H;
static constexpr size_t OFF_GU   = OFF_UP   + SZ_BIG_H;
static constexpr size_t OFF_BROW = OFF_GU   + SZ_BIG_H;

static inline void* wsp(void* ws, size_t off) { return (void*)((char*)ws + off); }

extern "C" void kernel_launch(void* const* d_in, const int* in_sizes, int n_in,
                              void* d_out, int out_size, void* d_ws, size_t ws_size,
                              hipStream_t stream) {
  (void)in_sizes; (void)n_in; (void)out_size; (void)ws_size;

  const float* x       = (const float*)d_in[0];
  const float* wq      = (const float*)d_in[1];
  const float* wk      = (const float*)d_in[2];
  const float* wv      = (const float*)d_in[3];
  const float* wo      = (const float*)d_in[4];
  const float* rel_pos = (const float*)d_in[5];
  const float* ln1_g   = (const float*)d_in[6];
  const float* ln1_b   = (const float*)d_in[7];
  const float* gate_w  = (const float*)d_in[8];
  const float* up_w    = (const float*)d_in[9];
  const float* down_w  = (const float*)d_in[10];
  const float* ln2_g   = (const float*)d_in[11];
  const float* ln2_b   = (const float*)d_in[12];

  _Float16* Xh   = (_Float16*)wsp(d_ws, OFF_XH);
  _Float16* Wt   = (_Float16*)wsp(d_ws, OFF_WT);
  _Float16* Qh   = (_Float16*)wsp(d_ws, OFF_QH);
  _Float16* Kh   = (_Float16*)wsp(d_ws, OFF_KH);
  _Float16* Vh   = (_Float16*)wsp(d_ws, OFF_VH);
  _Float16* Ctxh = (_Float16*)wsp(d_ws, OFF_CTX);
  float*    F32a = (float*)   wsp(d_ws, OFF_F32A);
  float*    H1   = (float*)   wsp(d_ws, OFF_H1);
  _Float16* H1h  = (_Float16*)wsp(d_ws, OFF_H1H);
  _Float16* Gh   = (_Float16*)wsp(d_ws, OFF_GATE);
  _Float16* Uh   = (_Float16*)wsp(d_ws, OFF_UP);
  _Float16* GUh  = (_Float16*)wsp(d_ws, OFF_GU);
  float*    Brow = (float*)   wsp(d_ws, OFF_BROW);

  const dim3 tb32x8(32, 8);
  const dim3 trD_D(Dq / 32, Dq / 32);    // 2048x2048 weight
  const dim3 trD_I(Iq / 32, Dq / 32);    // K=2048, N=8192
  const dim3 trI_D(Dq / 32, Iq / 32);    // K=8192, N=2048
  const dim3 gemmDD(Dq / 128, Mrows / 128);  // N=2048
  const dim3 gemmDI(Iq / 128, Mrows / 128);  // N=8192

  // Activations -> f16; bias row table
  cvt_f16_kernel<<<(Mrows * Dq) / 256, 256, 0, stream>>>(x, Xh, Mrows * Dq);
  bias_row_kernel<<<RELq, 32, 0, stream>>>(rel_pos, Brow);

  // Q = Xh @ wq  (f16 out)
  transpose_w_kernel<<<trD_D, tb32x8, 0, stream>>>(wq, Wt, Dq, Dq);
  gemm_wmma_kernel<true><<<gemmDD, 256, 0, stream>>>(Xh, Wt, nullptr, Qh, Mrows, Dq, Dq);
  // K
  transpose_w_kernel<<<trD_D, tb32x8, 0, stream>>>(wk, Wt, Dq, Dq);
  gemm_wmma_kernel<true><<<gemmDD, 256, 0, stream>>>(Xh, Wt, nullptr, Kh, Mrows, Dq, Dq);
  // V
  transpose_w_kernel<<<trD_D, tb32x8, 0, stream>>>(wv, Wt, Dq, Dq);
  gemm_wmma_kernel<true><<<gemmDD, 256, 0, stream>>>(Xh, Wt, nullptr, Vh, Mrows, Dq, Dq);

  // Attention (flash-style, WMMA)
  attn_kernel<<<Bq * Hq, 128, 0, stream>>>(Qh, Kh, Vh, Brow, Ctxh);

  // attn_out = ctx @ wo  (fp32 out)
  transpose_w_kernel<<<trD_D, tb32x8, 0, stream>>>(wo, Wt, Dq, Dq);
  gemm_wmma_kernel<false><<<gemmDD, 256, 0, stream>>>(Ctxh, Wt, F32a, nullptr, Mrows, Dq, Dq);

  // h1 = LN(x + attn_out), also f16 copy
  add_ln_kernel<<<Mrows, 256, 0, stream>>>(x, F32a, ln1_g, ln1_b, H1, H1h);

  // gate = h1 @ gate_w ; up = h1 @ up_w  (f16 out)
  transpose_w_kernel<<<trD_I, tb32x8, 0, stream>>>(gate_w, Wt, Dq, Iq);
  gemm_wmma_kernel<true><<<gemmDI, 256, 0, stream>>>(H1h, Wt, nullptr, Gh, Mrows, Iq, Dq);
  transpose_w_kernel<<<trD_I, tb32x8, 0, stream>>>(up_w, Wt, Dq, Iq);
  gemm_wmma_kernel<true><<<gemmDI, 256, 0, stream>>>(H1h, Wt, nullptr, Uh, Mrows, Iq, Dq);

  // gu = silu(gate) * up
  silu_mul_kernel<<<(Mrows * Iq) / 256, 256, 0, stream>>>(Gh, Uh, GUh, Mrows * Iq);

  // ffn = gu @ down_w  (fp32 out)
  transpose_w_kernel<<<trI_D, tb32x8, 0, stream>>>(down_w, Wt, Iq, Dq);
  gemm_wmma_kernel<false><<<gemmDD, 256, 0, stream>>>(GUh, Wt, F32a, nullptr, Mrows, Dq, Iq);

  // out = LN(h1 + ffn)
  add_ln_kernel<<<Mrows, 256, 0, stream>>>(H1, F32a, ln2_g, ln2_b, (float*)d_out, nullptr);
}
